// Agent_12489764897159
// MI455X (gfx1250) — matrix operand order repaired and checked by reading, not compiled, not run on hardware
//
#include <hip/hip_runtime.h>
#include <hip/hip_bf16.h>

typedef _Float16 h8  __attribute__((ext_vector_type(8)));
typedef _Float16 h16 __attribute__((ext_vector_type(16)));
typedef float    f8  __attribute__((ext_vector_type(8)));

constexpr int kOBS = 194;
constexpr int kKP  = 224;   // OBS padded up to a multiple of 32 (7 K-chunks)
constexpr int kH   = 64;
constexpr int kA   = 5;
constexpr int kE   = 3;
constexpr int kBM  = 64;    // rows per block (4 waves x 16)
constexpr int kFS  = 72;    // padded feat row stride in halves (144B, 16B-aligned)

__device__ __forceinline__ h16 cat16(h8 lo, h8 hi) {
  return __builtin_shufflevector(lo, hi, 0,1,2,3,4,5,6,7,8,9,10,11,12,13,14,15);
}
// A-fragment for 16-bit 16x32: lane holds 16 contiguous K-halves at p (K k0..k0+15)
__device__ __forceinline__ h16 ldA(const _Float16* p) {
  return cat16(*(const h8*)p, *(const h8*)(p + 8));
}
// B-fragment: lane holds 16 contiguous K-halves of one column (weights stored transposed)
__device__ __forceinline__ h16 ldB(const _Float16* p) {
  return cat16(*(const h8*)p, *(const h8*)(p + 8));
}

// Branch-free tanh: 1 - 2/(exp(2x)+1). Saturates to +/-1 on exp overflow/underflow,
// uses hardware v_exp_f32 / v_rcp_f32 (~6 VALU ops, no divergence) instead of
// the ~30-op branchy ocml tanhf path.
__device__ __forceinline__ float fast_tanh(float x) {
  float e = __expf(2.0f * x);
  return 1.0f - 2.0f * __builtin_amdgcn_rcpf(e + 1.0f);
}

__global__ __launch_bounds__(128, 1)
void Agent_12489764897159_kernel(
    const float* __restrict__ xg,  const int* __restrict__ act,
    const float* __restrict__ W1,  const float* __restrict__ b1,
    const float* __restrict__ W2,  const float* __restrict__ b2,
    const float* __restrict__ Wh,  const float* __restrict__ bh,
    const float* __restrict__ Wc1, const float* __restrict__ bc1,
    const float* __restrict__ Wc2, const float* __restrict__ bc2,
    const float* __restrict__ Wc3, const float* __restrict__ bc3,
    float* __restrict__ out, int Btot)
{
  __shared__ __align__(16) _Float16 sX [kBM * kKP];   // x tile, f16, K-padded
  __shared__ __align__(16) _Float16 sA1[kH * kKP];    // W1^T  [n][k]
  __shared__ __align__(16) _Float16 sC1[kH * kKP];    // Wc1^T [n][k]
  __shared__ __align__(16) _Float16 sA2[kH * kH];     // W2^T
  __shared__ __align__(16) _Float16 sC2[kH * kH];     // Wc2^T
  __shared__ __align__(16) _Float16 sF [4][16 * kFS]; // per-wave feat scratch
  __shared__ float sWh[kE * kH * kA];
  __shared__ float sBh[kE * kA];
  __shared__ float sB1[kH], sB2[kH], sBc1[kH], sBc2[kH], sV3[kH];
  __shared__ float sBc3v;

  const int tid  = threadIdx.x;
  const int row0 = blockIdx.x * kBM;

  // ---------- cooperative stage: x tile + all weights -> LDS ----------
  for (int i = tid; i < kBM * kKP; i += 128) {
    int r = i / kKP, k = i % kKP;
    int row = row0 + r;
    float v = (k < kOBS && row < Btot) ? xg[(size_t)row * kOBS + k] : 0.f;
    sX[i] = (_Float16)v;
  }
  for (int i = tid; i < kH * kKP; i += 128) {
    int n = i / kKP, k = i % kKP;
    sA1[i] = (_Float16)((k < kOBS) ? W1 [k * kH + n] : 0.f);
    sC1[i] = (_Float16)((k < kOBS) ? Wc1[k * kH + n] : 0.f);
  }
  for (int i = tid; i < kH * kH; i += 128) {
    int n = i / kH, k = i % kH;
    sA2[i] = (_Float16)W2 [k * kH + n];
    sC2[i] = (_Float16)Wc2[k * kH + n];
  }
  for (int i = tid; i < kE * kH * kA; i += 128) sWh[i] = Wh[i];
  if (tid < kE * kA) sBh[tid] = bh[tid];
  if (tid < kH) {
    sB1[tid] = b1[tid]; sB2[tid] = b2[tid];
    sBc1[tid] = bc1[tid]; sBc2[tid] = bc2[tid];
    sV3[tid] = Wc3[tid];
  }
  if (tid == 0) sBc3v = bc3[0];
  __syncthreads();

  const int lane = tid & 31;
  const int wave = tid >> 5;
  const int m    = lane & 15;   // sample-row (A/M) or column (B/C N) within tile
  const int hi   = lane >> 4;

  const _Float16* xrow = &sX[(wave * 16 + m) * kKP];
  _Float16* fW = &sF[wave][0];

  // x A-fragments: loaded once, reused by BOTH actor and critic layer 1
  h16 aX[7];
  #pragma unroll
  for (int kc = 0; kc < 7; ++kc) aX[kc] = ldA(xrow + kc * 32 + hi * 16);

  // ===================== actor layer 1: tanh(x@W1+b1) =====================
  #pragma unroll
  for (int nt = 0; nt < 4; ++nt) {
    f8 c = {0.f,0.f,0.f,0.f,0.f,0.f,0.f,0.f};
    #pragma unroll
    for (int kc = 0; kc < 7; ++kc) {
      h16 b = ldB(&sA1[(nt * 16 + m) * kKP + kc * 32 + hi * 16]);
      c = __builtin_amdgcn_wmma_f32_16x16x32_f16(false, aX[kc], false, b, (short)0, c, false, false);
    }
    float bias = sB1[nt * 16 + m];
    #pragma unroll
    for (int j = 0; j < 8; ++j)
      fW[(j + hi * 8) * kFS + nt * 16 + m] = (_Float16)fast_tanh(c[j] + bias);
  }

  // ===================== actor layer 2: tanh(h@W2+b2) =====================
  h16 a2[2];
  #pragma unroll
  for (int kc = 0; kc < 2; ++kc) a2[kc] = ldA(fW + m * kFS + kc * 32 + hi * 16);
  #pragma unroll
  for (int nt = 0; nt < 4; ++nt) {
    f8 c = {0.f,0.f,0.f,0.f,0.f,0.f,0.f,0.f};
    #pragma unroll
    for (int kc = 0; kc < 2; ++kc) {
      h16 b = ldB(&sA2[(nt * 16 + m) * kH + kc * 32 + hi * 16]);
      c = __builtin_amdgcn_wmma_f32_16x16x32_f16(false, a2[kc], false, b, (short)0, c, false, false);
    }
    float bias = sB2[nt * 16 + m];
    #pragma unroll
    for (int j = 0; j < 8; ++j)
      fW[(j + hi * 8) * kFS + nt * 16 + m] = (_Float16)fast_tanh(c[j] + bias);
  }

  // ====== routed head + log-softmax + entropy (lanes 0..15, one row each) ======
  {
    const int row = row0 + wave * 16 + m;
    if (hi == 0 && row < Btot) {
      const float* xr = xg + (size_t)row * kOBS;
      float v0 = xr[0], v1 = xr[1], v2 = xr[2];
      int ev = 0; float bv = v0;
      if (v1 > bv) { bv = v1; ev = 1; }
      if (v2 > bv) { ev = 2; }

      float lg[kA];
      #pragma unroll
      for (int a = 0; a < kA; ++a) lg[a] = sBh[ev * kA + a];
      const float* wh = &sWh[ev * kH * kA];
      #pragma unroll
      for (int hc = 0; hc < 8; ++hc) {
        h8 fv = *(const h8*)(fW + m * kFS + hc * 8);   // ds_load_b128
        #pragma unroll
        for (int t = 0; t < 8; ++t) {
          float f = (float)fv[t];
          int   h = hc * 8 + t;
          #pragma unroll
          for (int a = 0; a < kA; ++a) lg[a] += f * wh[h * kA + a];
        }
      }
      float mx = lg[0];
      #pragma unroll
      for (int a = 1; a < kA; ++a) mx = fmaxf(mx, lg[a]);
      float s = 0.f;
      #pragma unroll
      for (int a = 0; a < kA; ++a) s += __expf(lg[a] - mx);
      float lse = mx + __logf(s);
      int   an  = act[row];
      float ent = 0.f;
      #pragma unroll
      for (int a = 0; a < kA; ++a) {
        float lp = lg[a] - lse;
        ent -= __expf(lp) * lp;
      }
      out[row]             = (float)an;         // action passthrough
      out[Btot + row]      = lg[an] - lse;      // logp
      out[2 * Btot + row]  = ent;               // entropy
    }
  }

  // ===================== critic layer 1 (reuses aX fragments) =====================
  #pragma unroll
  for (int nt = 0; nt < 4; ++nt) {
    f8 c = {0.f,0.f,0.f,0.f,0.f,0.f,0.f,0.f};
    #pragma unroll
    for (int kc = 0; kc < 7; ++kc) {
      h16 b = ldB(&sC1[(nt * 16 + m) * kKP + kc * 32 + hi * 16]);
      c = __builtin_amdgcn_wmma_f32_16x16x32_f16(false, aX[kc], false, b, (short)0, c, false, false);
    }
    float bias = sBc1[nt * 16 + m];
    #pragma unroll
    for (int j = 0; j < 8; ++j)
      fW[(j + hi * 8) * kFS + nt * 16 + m] = (_Float16)fast_tanh(c[j] + bias);
  }

  // ===================== critic layer 2 =====================
  h16 a2c[2];
  #pragma unroll
  for (int kc = 0; kc < 2; ++kc) a2c[kc] = ldA(fW + m * kFS + kc * 32 + hi * 16);
  #pragma unroll
  for (int nt = 0; nt < 4; ++nt) {
    f8 c = {0.f,0.f,0.f,0.f,0.f,0.f,0.f,0.f};
    #pragma unroll
    for (int kc = 0; kc < 2; ++kc) {
      h16 b = ldB(&sC2[(nt * 16 + m) * kH + kc * 32 + hi * 16]);
      c = __builtin_amdgcn_wmma_f32_16x16x32_f16(false, a2c[kc], false, b, (short)0, c, false, false);
    }
    float bias = sBc2[nt * 16 + m];
    #pragma unroll
    for (int j = 0; j < 8; ++j)
      fW[(j + hi * 8) * kFS + nt * 16 + m] = (_Float16)fast_tanh(c[j] + bias);
  }

  // ===================== critic value head =====================
  {
    const int row = row0 + wave * 16 + m;
    if (hi == 0 && row < Btot) {
      float v = sBc3v;
      #pragma unroll
      for (int hc = 0; hc < 8; ++hc) {
        h8 fv = *(const h8*)(fW + m * kFS + hc * 8);   // ds_load_b128
        #pragma unroll
        for (int t = 0; t < 8; ++t)
          v += (float)fv[t] * sV3[hc * 8 + t];
      }
      out[3 * Btot + row] = v;
    }
  }
}

extern "C" void kernel_launch(void* const* d_in, const int* in_sizes, int n_in,
                              void* d_out, int out_size, void* d_ws, size_t ws_size,
                              hipStream_t stream) {
  const float* x   = (const float*)d_in[0];
  const int*   act = (const int*)  d_in[1];
  const float* W1  = (const float*)d_in[2];
  const float* b1  = (const float*)d_in[3];
  const float* W2  = (const float*)d_in[4];
  const float* b2  = (const float*)d_in[5];
  const float* Wh  = (const float*)d_in[6];
  const float* bh  = (const float*)d_in[7];
  const float* Wc1 = (const float*)d_in[8];
  const float* bc1 = (const float*)d_in[9];
  const float* Wc2 = (const float*)d_in[10];
  const float* bc2 = (const float*)d_in[11];
  const float* Wc3 = (const float*)d_in[12];
  const float* bc3 = (const float*)d_in[13];
  float* out = (float*)d_out;

  const int Btot = in_sizes[1];                 // action has B elements
  const int grid = (Btot + kBM - 1) / kBM;
  Agent_12489764897159_kernel<<<dim3(grid), dim3(128), 0, stream>>>(
      x, act, W1, b1, W2, b2, Wh, bh, Wc1, bc1, Wc2, bc2, Wc3, bc3, out, Btot);
}
